// ISTRModel_35424890258073
// MI455X (gfx1250) — matrix operand order repaired and checked by reading, not compiled, not run on hardware
//
#include <hip/hip_runtime.h>
#include <math.h>
#include <stdint.h>

// ---------------- model constants ----------------
constexpr int kB  = 16;
constexpr int kS  = 512;
constexpr int kD  = 1024;
constexpr int kH  = 16;
constexpr int kDH = 64;
constexpr int kFF = 4096;
constexpr int kRows = kB * kS;       // 8192
constexpr int kBand = 128;           // seq_len / sparsity_factor

typedef __attribute__((ext_vector_type(16))) __bf16 v16bf;
typedef __attribute__((ext_vector_type(8)))  float  v8f;
typedef __attribute__((ext_vector_type(4)))  int    v4i;

// async global->LDS (CDNA5); guarded so compilation can never break
#if defined(__HIP_DEVICE_COMPILE__) && __has_builtin(__builtin_amdgcn_global_load_async_to_lds_b128)
#define USE_ASYNC_LDS 1
#else
#define USE_ASYNC_LDS 0
#endif

__device__ __forceinline__ void g2l_16B(void* dst_lds, const void* src) {
#if USE_ASYNC_LDS
  __builtin_amdgcn_global_load_async_to_lds_b128(
      (__attribute__((address_space(1))) v4i*)(uintptr_t)src,
      (__attribute__((address_space(3))) v4i*)(unsigned)(uintptr_t)dst_lds,
      0, 0);
#else
  *(uint4*)dst_lds = *(const uint4*)src;
#endif
}

__device__ __forceinline__ void async_wait0() {
#if USE_ASYNC_LDS
#if __has_builtin(__builtin_amdgcn_s_wait_asynccnt)
  __builtin_amdgcn_s_wait_asynccnt(0);
#else
  asm volatile("s_wait_asynccnt 0" ::: "memory");
#endif
#endif
}

// ---------------- fragment slot maps (CDNA5 WMMA VGPR layouts) ----------------
// A (16x32 bf16): lane<16 holds row r, K in {0..7,16..23}; lane>=16 row r, K in {8..15,24..31}
__device__ __forceinline__ int a_slot(int r, int k) {
  int lane = ((k >> 3) & 1) ? (r + 16) : r;
  int e = (k & 7) | ((k & 16) ? 8 : 0);
  return lane * 16 + e;
}
// B (32x16 bf16): lane<16 = col c holds K {0..7,16..23}; lane>=16 = col c holds K {8..15,24..31}
__device__ __forceinline__ int b_slot(int k, int c) {
  return ((k & 8) ? (c + 16) : c) * 16 + ((k & 7) | ((k & 16) ? 8 : 0));
}

__device__ __forceinline__ v8f wmma_bf16(v16bf a, v16bf b, v8f c) {
  return __builtin_amdgcn_wmma_f32_16x16x32_bf16(false, a, false, b, (short)0, c,
                                                 false, false);
}

// =====================================================================
// weight pre-swizzle: W[K,N] fp32 -> bf16 fragment-major
//   frag f = (k/32)*(N/16) + (c/16), 512 halves each, slot = b_slot(k&31,c&15)
// thread: fixed column c, 4 consecutive k -> 4 contiguous bf16 slots (b64 store)
// =====================================================================
__global__ __launch_bounds__(256) void k_wconv(const float* __restrict__ W,
                                               __bf16* __restrict__ Wsw, int N) {
  const int c  = blockIdx.x * 256 + threadIdx.x;
  const int k0 = blockIdx.y * 4;
  float v[4];
#pragma unroll
  for (int j = 0; j < 4; ++j) v[j] = W[(size_t)(k0 + j) * N + c];
  __bf16* dst = Wsw + ((size_t)(k0 >> 5) * (N >> 4) + (c >> 4)) * 512 +
                (((k0 & 8) ? (c & 15) + 16 : (c & 15)) * 16) + ((k0 & 7) | ((k0 & 16) ? 8 : 0));
#pragma unroll
  for (int j = 0; j < 4; ++j) dst[j] = (__bf16)v[j];
}

// =====================================================================
// GEMM: Y[M,N] = act(X[M,K] * W[K,N] + bias)
//   X fp32 (converted to bf16 during staging), W pre-swizzled bf16.
//   tile 128x128x32, 8 waves (2x4), 8 WMMA accs/wave.
//   B-tile = contiguous 8KB async copy global->LDS.
// =====================================================================
__global__ __launch_bounds__(256) void k_gemm(const float* __restrict__ X,
                                              const __bf16* __restrict__ Wsw,
                                              const float* __restrict__ bias,
                                              float* __restrict__ Y,
                                              int N, int K, int act) {
  __shared__ __bf16 sA[8 * 512];
  __shared__ __bf16 sB[8 * 512];

  const int tid  = threadIdx.x;
  const int lane = tid & 31;
  const int wid  = tid >> 5;
  const int wm   = wid >> 2;
  const int wn   = wid & 3;
  const int m0   = blockIdx.y * 128;
  const int n0   = blockIdx.x * 128;

  const v8f vzero = {0.f, 0.f, 0.f, 0.f, 0.f, 0.f, 0.f, 0.f};
  v8f acc[4][2];
#pragma unroll
  for (int i = 0; i < 4; ++i)
#pragma unroll
    for (int j = 0; j < 2; ++j) acc[i][j] = vzero;

  const int ksteps = K >> 5;
  for (int kb = 0; kb < ksteps; ++kb) {
    // ---- B tile: contiguous 8KB copy (async global->LDS on CDNA5)
    const __bf16* wsrc = Wsw + ((size_t)kb * (N >> 4) + (n0 >> 4)) * 512;
#pragma unroll
    for (int i = 0; i < 2; ++i) {
      int ch = i * 256 + tid;
      g2l_16B(sB + ch * 8, wsrc + ch * 8);
    }
    // ---- A tile: batch loads, then fp32->bf16 swizzled stores (merge to b64)
    float4 va[4];
#pragma unroll
    for (int i = 0; i < 4; ++i) {
      int base = (i * 256 + tid) * 4;
      int r = base >> 5, k = base & 31;
      va[i] = *(const float4*)(X + (size_t)(m0 + r) * K + kb * 32 + k);
    }
#pragma unroll
    for (int i = 0; i < 4; ++i) {
      int base = (i * 256 + tid) * 4;
      int r = base >> 5, k = base & 31;
      __bf16* fb = sA + (r >> 4) * 512;
      int rr = r & 15;
      fb[a_slot(rr, k + 0)] = (__bf16)va[i].x;
      fb[a_slot(rr, k + 1)] = (__bf16)va[i].y;
      fb[a_slot(rr, k + 2)] = (__bf16)va[i].z;
      fb[a_slot(rr, k + 3)] = (__bf16)va[i].w;
    }
    if (kb + 1 < ksteps) {
      __builtin_prefetch(X + (size_t)(m0 + (tid >> 1)) * K + (kb + 1) * 32 + ((tid & 1) << 4), 0, 3);
      __builtin_prefetch(Wsw + ((size_t)(kb + 1) * (N >> 4) + (n0 >> 4)) * 512 + tid * 16, 0, 3);
    }
    async_wait0();
    __syncthreads();

#pragma unroll
    for (int mt = 0; mt < 4; ++mt) {
      v16bf af = *(const v16bf*)(sA + (wm * 4 + mt) * 512 + lane * 16);
#pragma unroll
      for (int nt = 0; nt < 2; ++nt) {
        v16bf bf = *(const v16bf*)(sB + (wn * 2 + nt) * 512 + lane * 16);
        acc[mt][nt] = wmma_bf16(af, bf, acc[mt][nt]);
      }
    }
    __syncthreads();
  }

  const int rbase = m0 + wm * 64 + ((lane & 16) ? 8 : 0);
  const int cl = lane & 15;
#pragma unroll
  for (int mt = 0; mt < 4; ++mt) {
#pragma unroll
    for (int nt = 0; nt < 2; ++nt) {
      int col = n0 + wn * 32 + nt * 16 + cl;
      float bv = bias[col];
#pragma unroll
      for (int r = 0; r < 8; ++r) {
        float v = acc[mt][nt][r] + bv;
        if (act == 1) v = 0.5f * v * (1.f + erff(v * 0.70710678118654752f));
        Y[(size_t)(rbase + mt * 16 + r) * N + col] = v;
      }
    }
  }
}

// =====================================================================
// Fused band-masked attention (one block per (b, h, 64-row q-block))
// =====================================================================
constexpr int SMEM_S  = 0;                     // 64x512 fp32 scores  : 131072 B
constexpr int SMEM_P  = 131072;                // 64 P fragments      :  65536 B
constexpr int SMEM_Q  = 131072 + 65536;        // 8 Q fragments       :   8192 B
constexpr int SMEM_KV = 131072 + 65536 + 8192; // 8 K/V fragments     :   8192 B
constexpr int SMEM_TOTAL = SMEM_KV + 8192;     // 212992 B

__global__ __launch_bounds__(256) void k_attn(const float* __restrict__ Qb,
                                              const float* __restrict__ Kb,
                                              const float* __restrict__ Vb,
                                              float* __restrict__ Ob) {
  extern __shared__ char smem[];
  float*  sS  = (float*)(smem + SMEM_S);
  __bf16* sP  = (__bf16*)(smem + SMEM_P);
  __bf16* sQ  = (__bf16*)(smem + SMEM_Q);
  __bf16* sKV = (__bf16*)(smem + SMEM_KV);

  const int tid = threadIdx.x, lane = tid & 31, wid = tid >> 5;
  int bid = blockIdx.x;
  const int qb = bid & 7;  bid >>= 3;
  const int h  = bid & 15; bid >>= 4;
  const int b  = bid;
  const int q0 = qb * 64;
  const size_t rowbase = ((size_t)b * kS) * kD + h * kDH;

  // ---- stage Q block (64x64) as A fragments [qt(4)][kg(2)]
  {
    float4 va[4];
#pragma unroll
    for (int i = 0; i < 4; ++i) {
      int base = (i * 256 + tid) * 4;
      int r = base >> 6, c = base & 63;
      va[i] = *(const float4*)(Qb + rowbase + (size_t)(q0 + r) * kD + c);
    }
#pragma unroll
    for (int i = 0; i < 4; ++i) {
      int base = (i * 256 + tid) * 4;
      int r = base >> 6, c = base & 63;
      __bf16* fb = sQ + ((r >> 4) * 2 + (c >> 5)) * 512;
      int rr = r & 15, kk = c & 31;
      fb[a_slot(rr, kk + 0)] = (__bf16)va[i].x;
      fb[a_slot(rr, kk + 1)] = (__bf16)va[i].y;
      fb[a_slot(rr, kk + 2)] = (__bf16)va[i].z;
      fb[a_slot(rr, kk + 3)] = (__bf16)va[i].w;
    }
  }
  // ---- init scores to mask value
  float4 neg = make_float4(-1e9f, -1e9f, -1e9f, -1e9f);
  for (int i = tid; i < 64 * 512 / 4; i += 256) ((float4*)sS)[i] = neg;
  __syncthreads();

  int lo = q0 - kBand;       if (lo < 0) lo = 0;
  int hi = q0 + 63 + kBand;  if (hi > kS - 1) hi = kS - 1;
  const int kb_lo = lo >> 6, kb_hi = hi >> 6;

  // ================= scores: S = Q K^T / 8, banded =================
  for (int kbk = kb_lo; kbk <= kb_hi; ++kbk) {
    // stage K block as B frags for Q*K^T: [kg(dh/32)][ct(4)]
    {
      float4 va[4];
#pragma unroll
      for (int i = 0; i < 4; ++i) {
        int base = (i * 256 + tid) * 4;
        int r = base >> 6, c = base & 63;  // r = key row in block, c = dh
        va[i] = *(const float4*)(Kb + rowbase + (size_t)(kbk * 64 + r) * kD + c);
      }
#pragma unroll
      for (int i = 0; i < 4; ++i) {
        int base = (i * 256 + tid) * 4;
        int r = base >> 6, c = base & 63;
        __bf16* fb = sKV + ((c >> 5) * 4 + (r >> 4)) * 512;
        int cc = r & 15, kk = c & 31;
        // contiguous e-run: same lane, e = (kk&7)|(kk&16?8:0) .. +3
        fb[b_slot(kk + 0, cc)] = (__bf16)va[i].x;
        fb[b_slot(kk + 1, cc)] = (__bf16)va[i].y;
        fb[b_slot(kk + 2, cc)] = (__bf16)va[i].z;
        fb[b_slot(kk + 3, cc)] = (__bf16)va[i].w;
      }
    }
    __syncthreads();
    const v8f vz = {0.f, 0.f, 0.f, 0.f, 0.f, 0.f, 0.f, 0.f};
#pragma unroll
    for (int u = 0; u < 2; ++u) {
      int tt = wid * 2 + u;
      int qt = tt >> 2, ct = tt & 3;
      v8f acc = vz;
#pragma unroll
      for (int kg = 0; kg < 2; ++kg) {
        v16bf a = *(const v16bf*)(sQ + (qt * 2 + kg) * 512 + lane * 16);
        v16bf bb = *(const v16bf*)(sKV + (kg * 4 + ct) * 512 + lane * 16);
        acc = wmma_bf16(a, bb, acc);
      }
      int ib = qt * 16 + ((lane & 16) ? 8 : 0);
      int j  = kbk * 64 + ct * 16 + (lane & 15);
#pragma unroll
      for (int r = 0; r < 8; ++r) {
        int iloc = ib + r;
        int d = (q0 + iloc) - j;
        float v = (d >= -kBand && d <= kBand) ? acc[r] * 0.125f : -1e9f;
        sS[iloc * 512 + j] = v;
      }
    }
    __syncthreads();
  }

  // ================= softmax rows -> bf16 P fragments =================
  // lane owns 16 *contiguous* columns [lane*16, lane*16+16): vectorized LDS
  // reads, and P writes collapse to two contiguous 16B runs per lane.
  for (int rr = 0; rr < 8; ++rr) {
    int iloc = wid * 8 + rr;
    const float4* src = (const float4*)(sS + iloc * 512 + lane * 16);
    float vals[16];
    float mx = -3.0e38f;
#pragma unroll
    for (int q = 0; q < 4; ++q) {
      float4 v4 = src[q];
      vals[q * 4 + 0] = v4.x; vals[q * 4 + 1] = v4.y;
      vals[q * 4 + 2] = v4.z; vals[q * 4 + 3] = v4.w;
    }
#pragma unroll
    for (int t = 0; t < 16; ++t) mx = fmaxf(mx, vals[t]);
#pragma unroll
    for (int off = 16; off > 0; off >>= 1) mx = fmaxf(mx, __shfl_xor(mx, off));
    float sum = 0.f;
#pragma unroll
    for (int t = 0; t < 16; ++t) { vals[t] = __expf(vals[t] - mx); sum += vals[t]; }
#pragma unroll
    for (int off = 16; off > 0; off >>= 1) sum += __shfl_xor(sum, off);
    float inv = 1.f / sum;
    int qt = iloc >> 4, r16 = iloc & 15;
    __bf16* dst = sP + (qt * 16 + (lane >> 1)) * 512;  // kg = lane/2
    int ebase = (lane & 1) * 8;                        // k-in-group base 0 or 16
#pragma unroll
    for (int j = 0; j < 8; ++j) dst[r16 * 16 + ebase + j] = (__bf16)(vals[j] * inv);
#pragma unroll
    for (int j = 0; j < 8; ++j) dst[(r16 + 16) * 16 + ebase + j] = (__bf16)(vals[8 + j] * inv);
  }
  __syncthreads();

  // ================= ctx = P * V (banded) =================
  const v8f vz2 = {0.f, 0.f, 0.f, 0.f, 0.f, 0.f, 0.f, 0.f};
  v8f o[2] = {vz2, vz2};
  const int tt0 = wid * 2;
  for (int kbk = kb_lo; kbk <= kb_hi; ++kbk) {
    // stage V block as B frags [kg(2)][dt(4)]; thread: fixed dh col, 4 key rows
    {
      float vv[4][4];
#pragma unroll
      for (int i = 0; i < 4; ++i) {
        int idx = i * 256 + tid;
        int c = idx & 63, kq = idx >> 6;  // c = dh col, kq = key-row quad
#pragma unroll
        for (int j = 0; j < 4; ++j)
          vv[i][j] = Vb[rowbase + (size_t)(kbk * 64 + kq * 4 + j) * kD + c];
      }
#pragma unroll
      for (int i = 0; i < 4; ++i) {
        int idx = i * 256 + tid;
        int c = idx & 63, kq = idx >> 6;
        int k0 = kq * 4;
        __bf16* dst = sKV + ((kq >> 3) * 4 + (c >> 4)) * 512 +
                      (((k0 & 8) ? (c & 15) + 16 : (c & 15)) * 16) + ((k0 & 7) | ((k0 & 16) ? 8 : 0));
#pragma unroll
        for (int j = 0; j < 4; ++j) dst[j] = (__bf16)vv[i][j];  // contiguous b64
      }
    }
    __syncthreads();
#pragma unroll
    for (int u = 0; u < 2; ++u) {
      int qt = (tt0 + u) >> 2, dt = (tt0 + u) & 3;
#pragma unroll
      for (int kg = 0; kg < 2; ++kg) {
        v16bf p  = *(const v16bf*)(sP + (qt * 16 + kbk * 2 + kg) * 512 + lane * 16);
        v16bf vv2 = *(const v16bf*)(sKV + (kg * 4 + dt) * 512 + lane * 16);
        o[u] = wmma_bf16(p, vv2, o[u]);
      }
    }
    __syncthreads();
  }
#pragma unroll
  for (int u = 0; u < 2; ++u) {
    int qt = (tt0 + u) >> 2, dt = (tt0 + u) & 3;
    int rb = q0 + qt * 16 + ((lane & 16) ? 8 : 0);
    int col = dt * 16 + (lane & 15);
#pragma unroll
    for (int r = 0; r < 8; ++r)
      Ob[rowbase + (size_t)(rb + r) * kD + col] = o[u][r];
  }
}

// =====================================================================
// y = LayerNorm(x + res) * g + b
// =====================================================================
__global__ __launch_bounds__(256) void k_add_ln(const float* __restrict__ X,
                                                const float* __restrict__ R,
                                                const float* __restrict__ g,
                                                const float* __restrict__ be,
                                                float* __restrict__ Y) {
  __shared__ float red[16];
  const int row = blockIdx.x, tid = threadIdx.x, lane = tid & 31, wid = tid >> 5;
  const size_t base = (size_t)row * kD;
  float v[4];
  float s1 = 0.f, s2 = 0.f;
#pragma unroll
  for (int i = 0; i < 4; ++i) {
    int d = tid + i * 256;
    float x = X[base + d] + R[base + d];
    v[i] = x; s1 += x; s2 += x * x;
  }
#pragma unroll
  for (int off = 16; off > 0; off >>= 1) { s1 += __shfl_xor(s1, off); s2 += __shfl_xor(s2, off); }
  if (lane == 0) { red[wid] = s1; red[8 + wid] = s2; }
  __syncthreads();
  if (wid == 0) {
    float a = (lane < 8) ? red[lane] : 0.f;
    float c = (lane < 8) ? red[8 + lane] : 0.f;
#pragma unroll
    for (int off = 4; off > 0; off >>= 1) { a += __shfl_xor(a, off); c += __shfl_xor(c, off); }
    if (lane == 0) { red[0] = a * (1.f / kD); red[1] = c * (1.f / kD); }
  }
  __syncthreads();
  float mean = red[0];
  float var  = red[1] - mean * mean;
  float rs   = rsqrtf(var + 1e-5f);
#pragma unroll
  for (int i = 0; i < 4; ++i) {
    int d = tid + i * 256;
    Y[base + d] = (v[i] - mean) * rs * g[d] + be[d];
  }
}

// =====================================================================
// embedding: y = (x @ W + b) * sqrt(D) + PE(s)
// =====================================================================
__global__ __launch_bounds__(256) void k_embed(const float* __restrict__ Xin,
                                               const float* __restrict__ W,
                                               const float* __restrict__ bias,
                                               float* __restrict__ Y, int Cin) {
  const int row = blockIdx.x;
  const int s = row & (kS - 1);
  const int tid = threadIdx.x;
  float xr[8];
  for (int c = 0; c < Cin; ++c) xr[c] = Xin[(size_t)row * Cin + c];
  const float pos = (float)s;
#pragma unroll
  for (int i = 0; i < 4; ++i) {
    int d = tid + i * 256;
    float a = bias[d];
    for (int c = 0; c < Cin; ++c) a += xr[c] * W[c * kD + d];
    a *= 32.0f;  // sqrt(1024)
    float freq = __expf((float)(d & ~1) * (-9.2103403719761836f / (float)kD));
    float sv, cv;
    __sincosf(pos * freq, &sv, &cv);
    a += (d & 1) ? cv : sv;
    Y[(size_t)row * kD + d] = a;
  }
}

// =====================================================================
// final: out[row] = dot(dec[row,:], out_w) + out_b
// =====================================================================
__global__ __launch_bounds__(256) void k_out(const float* __restrict__ Xd,
                                             const float* __restrict__ w,
                                             const float* __restrict__ b0,
                                             float* __restrict__ out) {
  __shared__ float red[8];
  const int row = blockIdx.x, tid = threadIdx.x, lane = tid & 31, wid = tid >> 5;
  float s = 0.f;
#pragma unroll
  for (int i = 0; i < 4; ++i) {
    int d = tid + i * 256;
    s += Xd[(size_t)row * kD + d] * w[d];
  }
#pragma unroll
  for (int off = 16; off > 0; off >>= 1) s += __shfl_xor(s, off);
  if (lane == 0) red[wid] = s;
  __syncthreads();
  if (tid == 0) {
    float t = 0.f;
    for (int i = 0; i < 8; ++i) t += red[i];
    out[row] = t + b0[0];
  }
}

// =====================================================================
// host orchestration
// =====================================================================
struct AttnP { const float *wq, *bq, *wk, *bk, *wv, *bv, *wo, *bo; };
struct EncP  { AttnP at; const float *l1w, *l1b, *l2w, *l2b, *g1, *b1, *g2, *b2; };
struct DecP  { AttnP sa, ca; const float *l1w, *l1b, *l2w, *l2b, *g1, *b1, *g2, *b2, *g3, *b3; };
struct AttnW { const __bf16 *wq, *wk, *wv, *wo; };

extern "C" void kernel_launch(void* const* d_in, const int* in_sizes, int n_in,
                              void* d_out, int out_size, void* d_ws, size_t ws_size,
                              hipStream_t stream) {
  (void)out_size; (void)ws_size;
  if (n_in < 66) return;
  auto F = [&](int i) { return (const float*)d_in[i]; };

  const float *x_enc, *x_dec, *enc_emb_w, *enc_emb_b, *dec_emb_w, *dec_emb_b, *out_w, *out_b;
  EncP enc[2];
  DecP dec;

  if (in_sizes[0] == kB * kS * 7) {
    // ---- recursive insertion order ----
    x_enc = F(0); x_dec = F(1);
    for (int L = 0; L < 2; ++L) {
      int o = 2 + L * 16;
      enc[L].l1w = F(o + 0); enc[L].l1b = F(o + 1); enc[L].l2w = F(o + 2); enc[L].l2b = F(o + 3);
      enc[L].g1 = F(o + 4); enc[L].b1 = F(o + 5); enc[L].g2 = F(o + 6); enc[L].b2 = F(o + 7);
      enc[L].at = {F(o + 8), F(o + 9), F(o + 10), F(o + 11), F(o + 12), F(o + 13), F(o + 14), F(o + 15)};
    }
    dec.l1w = F(34); dec.l1b = F(35); dec.l2w = F(36); dec.l2b = F(37);
    dec.g1 = F(38); dec.b1 = F(39); dec.g2 = F(40); dec.b2 = F(41); dec.g3 = F(42); dec.b3 = F(43);
    dec.sa = {F(44), F(45), F(46), F(47), F(48), F(49), F(50), F(51)};
    dec.ca = {F(52), F(53), F(54), F(55), F(56), F(57), F(58), F(59)};
    enc_emb_w = F(60); enc_emb_b = F(61); dec_emb_w = F(62); dec_emb_b = F(63);
    out_w = F(64); out_b = F(65);
  } else {
    // ---- jax pytree (sorted-key) order ----
    dec_emb_b = F(0); dec_emb_w = F(1);
    dec.ca = {F(8), F(4), F(6), F(2), F(9), F(5), F(7), F(3)};
    dec.l1b = F(10); dec.l1w = F(11); dec.l2b = F(12); dec.l2w = F(13);
    dec.b1 = F(14); dec.g1 = F(15); dec.b2 = F(16); dec.g2 = F(17); dec.b3 = F(18); dec.g3 = F(19);
    dec.sa = {F(26), F(22), F(24), F(20), F(27), F(23), F(25), F(21)};
    enc_emb_b = F(28); enc_emb_w = F(29);
    for (int L = 0; L < 2; ++L) {
      int o = 30 + L * 16;
      enc[L].at = {F(o + 6), F(o + 2), F(o + 4), F(o + 0), F(o + 7), F(o + 3), F(o + 5), F(o + 1)};
      enc[L].l1b = F(o + 8); enc[L].l1w = F(o + 9); enc[L].l2b = F(o + 10); enc[L].l2w = F(o + 11);
      enc[L].b1 = F(o + 12); enc[L].g1 = F(o + 13); enc[L].b2 = F(o + 14); enc[L].g2 = F(o + 15);
    }
    out_b = F(62); out_w = F(63); x_dec = F(64); x_enc = F(65);
  }

  // ---- workspace carving (floats): 11*32MiB activations + 80MB bf16 weights
  float* ws = (float*)d_ws;
  const size_t ACT = (size_t)kRows * kD;
  float* ENCb = ws + 0 * ACT;
  float* TMP  = ws + 1 * ACT;
  float* Qb   = ws + 2 * ACT;
  float* Kb2  = ws + 3 * ACT;
  float* Vb2  = ws + 4 * ACT;
  float* CTX  = ws + 5 * ACT;
  float* FFNb = ws + 6 * ACT;   // 4*ACT
  float* DECb = ws + 10 * ACT;
  __bf16* wpool = (__bf16*)(ws + 11 * ACT);
  size_t woff = 0;

  dim3 blk(256);
  auto conv = [&](const float* W, int K, int N) -> const __bf16* {
    __bf16* dst = wpool + woff;
    woff += (size_t)K * N;
    k_wconv<<<dim3(N / 256, K / 4), blk, 0, stream>>>(W, dst, N);
    return dst;
  };

  // pre-swizzle all WMMA weights to bf16 fragment-major
  AttnW encw[2], saw, caw;
  const __bf16 *encl1[2], *encl2[2], *decl1, *decl2;
  for (int L = 0; L < 2; ++L) {
    encw[L].wq = conv(enc[L].at.wq, kD, kD);
    encw[L].wk = conv(enc[L].at.wk, kD, kD);
    encw[L].wv = conv(enc[L].at.wv, kD, kD);
    encw[L].wo = conv(enc[L].at.wo, kD, kD);
    encl1[L]   = conv(enc[L].l1w, kD, kFF);
    encl2[L]   = conv(enc[L].l2w, kFF, kD);
  }
  saw.wq = conv(dec.sa.wq, kD, kD); saw.wk = conv(dec.sa.wk, kD, kD);
  saw.wv = conv(dec.sa.wv, kD, kD); saw.wo = conv(dec.sa.wo, kD, kD);
  caw.wq = conv(dec.ca.wq, kD, kD); caw.wk = conv(dec.ca.wk, kD, kD);
  caw.wv = conv(dec.ca.wv, kD, kD); caw.wo = conv(dec.ca.wo, kD, kD);
  decl1 = conv(dec.l1w, kD, kFF);
  decl2 = conv(dec.l2w, kFF, kD);

  auto gemm = [&](const float* X, const __bf16* Wsw, const float* bia, float* Y,
                  int N, int K, int act) {
    dim3 grid(N / 128, kRows / 128);
    k_gemm<<<grid, blk, 0, stream>>>(X, Wsw, bia, Y, N, K, act);
  };
  auto addln = [&](const float* X, const float* R, const float* g, const float* be, float* Y) {
    k_add_ln<<<dim3(kRows), blk, 0, stream>>>(X, R, g, be, Y);
  };
  auto attn = [&](const float* q, const float* k, const float* v, float* o) {
    k_attn<<<dim3(kB * kH * (kS / 64)), blk, SMEM_TOTAL, stream>>>(q, k, v, o);
  };

  // ---------------- encoder ----------------
  k_embed<<<dim3(kRows), blk, 0, stream>>>(x_enc, enc_emb_w, enc_emb_b, ENCb, 7);
  for (int L = 0; L < 2; ++L) {
    gemm(ENCb, encw[L].wq, enc[L].at.bq, Qb,  kD, kD, 0);
    gemm(ENCb, encw[L].wk, enc[L].at.bk, Kb2, kD, kD, 0);
    gemm(ENCb, encw[L].wv, enc[L].at.bv, Vb2, kD, kD, 0);
    attn(Qb, Kb2, Vb2, CTX);
    gemm(CTX, encw[L].wo, enc[L].at.bo, TMP, kD, kD, 0);
    addln(TMP, ENCb, enc[L].g1, enc[L].b1, ENCb);
    gemm(ENCb, encl1[L], enc[L].l1b, FFNb, kFF, kD, 1);
    gemm(FFNb, encl2[L], enc[L].l2b, TMP,  kD, kFF, 0);
    addln(TMP, ENCb, enc[L].g2, enc[L].b2, ENCb);
  }

  // ---------------- decoder ----------------
  k_embed<<<dim3(kRows), blk, 0, stream>>>(x_dec, dec_emb_w, dec_emb_b, DECb, 1);
  // self-attention
  gemm(DECb, saw.wq, dec.sa.bq, Qb,  kD, kD, 0);
  gemm(DECb, saw.wk, dec.sa.bk, Kb2, kD, kD, 0);
  gemm(DECb, saw.wv, dec.sa.bv, Vb2, kD, kD, 0);
  attn(Qb, Kb2, Vb2, CTX);
  gemm(CTX, saw.wo, dec.sa.bo, TMP, kD, kD, 0);
  addln(TMP, DECb, dec.g1, dec.b1, DECb);
  // cross-attention (K/V from encoder memory)
  gemm(DECb, caw.wq, dec.ca.bq, Qb,  kD, kD, 0);
  gemm(ENCb, caw.wk, dec.ca.bk, Kb2, kD, kD, 0);
  gemm(ENCb, caw.wv, dec.ca.bv, Vb2, kD, kD, 0);
  attn(Qb, Kb2, Vb2, CTX);
  gemm(CTX, caw.wo, dec.ca.bo, TMP, kD, kD, 0);
  addln(TMP, DECb, dec.g2, dec.b2, DECb);
  // FFN
  gemm(DECb, decl1, dec.l1b, FFNb, kFF, kD, 1);
  gemm(FFNb, decl2, dec.l2b, TMP,  kD, kFF, 0);
  addln(TMP, DECb, dec.g3, dec.b3, DECb);

  k_out<<<dim3(kRows), blk, 0, stream>>>(DECb, out_w, out_b, (float*)d_out);
}
